// DeepSeekMoE_57578331570801
// MI455X (gfx1250) — matrix operand order repaired and checked by reading, not compiled, not run on hardware
//
#include <hip/hip_runtime.h>

// ---------------- problem constants ----------------
constexpr int cB  = 2;
constexpr int cS  = 2048;
constexpr int cD  = 1024;   // d_model
constexpr int cE  = 16;     // experts
constexpr int cK  = 2;      // top-k
constexpr int cI  = 1024;   // expert intermediate
constexpr int cSI = 2048;   // shared intermediate
constexpr int cN  = cB * cS;

// ---------------- vector types for WMMA ----------------
typedef __attribute__((ext_vector_type(16))) __bf16 v16bf;
typedef __attribute__((ext_vector_type(8)))  __bf16 v8bf;
typedef __attribute__((ext_vector_type(4)))  __bf16 v4bf;
typedef __attribute__((ext_vector_type(8)))  float  v8f;

// ---------------- optional gfx1250 async -> LDS path ----------------
#if __has_builtin(__builtin_amdgcn_global_load_async_to_lds_b128) && \
    __has_builtin(__builtin_amdgcn_s_wait_asynccnt)
#define USE_ASYNC_LDS 1
typedef int v4i_async __attribute__((vector_size(16)));
typedef __attribute__((address_space(1))) v4i_async* gv4i_p;   // global (AS1)
typedef __attribute__((address_space(3))) v4i_async* lv4i_p;   // LDS (AS3)
__device__ inline gv4i_p to_global_v4(const void* p) {
    return (gv4i_p)(unsigned long long)(uintptr_t)p;   // generic global == AS1 bits
}
__device__ inline lv4i_p to_lds_v4(void* p) {
    return (lv4i_p)(unsigned int)(uintptr_t)p;         // LDS offset = low 32 bits
}
#endif

// =====================================================================
// Small helper kernels
// =====================================================================
__global__ void init_kernel(int* counts) {
    if (threadIdx.x < cE) counts[threadIdx.x] = 0;
}

__global__ void convert_x_kernel(const float* __restrict__ x, __bf16* __restrict__ xb) {
    size_t i = ((size_t)blockIdx.x * 256 + threadIdx.x) * 4;
    float4 v = *(const float4*)&x[i];
    v4bf o;
    o[0] = (__bf16)v.x; o[1] = (__bf16)v.y; o[2] = (__bf16)v.z; o[3] = (__bf16)v.w;
    *(v4bf*)&xb[i] = o;
}

// Transpose [R][C] f32 -> [C][R] bf16 (all dims multiples of 32), batched on z.
__global__ __launch_bounds__(256) void transpose_cvt_kernel(
    const float* __restrict__ src, __bf16* __restrict__ dst, int R, int C) {
    __shared__ float tile[32][33];
    size_t base = (size_t)blockIdx.z * R * C;
    src += base; dst += base;
    int c0 = blockIdx.x * 32, r0 = blockIdx.y * 32;
    int tx = threadIdx.x, ty = threadIdx.y;     // block (32,8)
    #pragma unroll
    for (int j = 0; j < 32; j += 8)
        tile[ty + j][tx] = src[(size_t)(r0 + ty + j) * C + (c0 + tx)];
    __syncthreads();
    #pragma unroll
    for (int j = 0; j < 32; j += 8)
        dst[(size_t)(c0 + ty + j) * R + (r0 + tx)] = (__bf16)tile[tx][ty + j];
}

// Router logits: one wave32 per token, 8 tokens per 256-thread block.
__global__ __launch_bounds__(256) void router_kernel(
    const float* __restrict__ x, const float* __restrict__ Wr, float* __restrict__ logits) {
    int lane = threadIdx.x & 31, wave = threadIdx.x >> 5;
    int n = blockIdx.x * 8 + wave;
    float acc[cE];
    #pragma unroll
    for (int e = 0; e < cE; ++e) acc[e] = 0.f;
    for (int d = lane; d < cD; d += 32) {
        float xv = x[(size_t)n * cD + d];
        #pragma unroll
        for (int e = 0; e < cE; ++e) acc[e] += xv * Wr[e * cD + d];
    }
    #pragma unroll
    for (int e = 0; e < cE; ++e)
        for (int off = 16; off > 0; off >>= 1) acc[e] += __shfl_xor(acc[e], off, 32);
    if (lane == 0) {
        #pragma unroll
        for (int e = 0; e < cE; ++e) logits[(size_t)n * cE + e] = acc[e];
    }
}

// Top-2 + softmax over selected; count tokens per expert.
__global__ void topk_kernel(const float* __restrict__ logits,
                            int* __restrict__ ae, float* __restrict__ aw,
                            int* __restrict__ counts) {
    int n = blockIdx.x * 256 + threadIdx.x;
    float l[cE];
    #pragma unroll
    for (int e = 0; e < cE; ++e) l[e] = logits[(size_t)n * cE + e];
    int b1 = 0; float v1 = l[0];
    #pragma unroll
    for (int e = 1; e < cE; ++e) if (l[e] > v1) { v1 = l[e]; b1 = e; }
    int b2 = 0; float v2 = -3.4e38f;
    #pragma unroll
    for (int e = 0; e < cE; ++e) if (e != b1 && l[e] > v2) { v2 = l[e]; b2 = e; }
    float w2 = __expf(v2 - v1);
    float inv = __builtin_amdgcn_rcpf(1.0f + w2);
    ae[n * 2 + 0] = b1; aw[n * 2 + 0] = inv;
    ae[n * 2 + 1] = b2; aw[n * 2 + 1] = w2 * inv;
    atomicAdd(&counts[b1], 1);
    atomicAdd(&counts[b2], 1);
}

__global__ void scan_kernel(const int* __restrict__ counts,
                            int* __restrict__ offsets, int* __restrict__ cursors) {
    if (threadIdx.x == 0) {
        int s = 0;
        for (int e = 0; e < cE; ++e) { offsets[e] = s; cursors[e] = s; s += counts[e]; }
        offsets[cE] = s;
    }
}

__global__ void scatter_kernel(const int* __restrict__ ae, const float* __restrict__ aw,
                               int* __restrict__ cursors,
                               int* __restrict__ btok, float* __restrict__ bw,
                               int* __restrict__ posof) {
    int id = blockIdx.x * 256 + threadIdx.x;       // assignment id in [0, N*K)
    int e = ae[id];
    int pos = atomicAdd(&cursors[e], 1);
    btok[pos] = id >> 1;
    bw[pos] = aw[id];
    posof[id] = pos;
}

// out += eo[pos(n,0)] + eo[pos(n,1)]  (deterministic, fixed order)
__global__ void combine_kernel(float* __restrict__ out, const float* __restrict__ eo,
                               const int* __restrict__ posof) {
    size_t idx = ((size_t)blockIdx.x * 256 + threadIdx.x) * 4;
    int n = (int)(idx >> 10);                      // / cD
    int col = (int)(idx & (cD - 1));
    int p0 = posof[n * 2 + 0], p1 = posof[n * 2 + 1];
    float4 a = *(const float4*)&eo[(size_t)p0 * cD + col];
    float4 b = *(const float4*)&eo[(size_t)p1 * cD + col];
    float4 o = *(float4*)&out[idx];
    o.x += a.x + b.x; o.y += a.y + b.y; o.z += a.z + b.z; o.w += a.w + b.w;
    *(float4*)&out[idx] = o;
}

// =====================================================================
// WMMA fragment loaders (wave32; A and B both use contiguous k = 16*(lane/16) + s)
// =====================================================================
// A (16x32 bf16) from row-major LDS tile. lane<16: M=lane, K = kb+0..15;
// lane>=16: M=lane-16, K = kb+16..31.
__device__ inline v16bf load_a_frag(const __bf16* __restrict__ base, int stride,
                                    int kb, int lane) {
    const __bf16* p = base + (size_t)(lane & 15) * stride + kb + ((lane < 16) ? 0 : 16);
    v16bf a;
    #pragma unroll
    for (int t = 0; t < 8; ++t) a[t] = p[t];
    #pragma unroll
    for (int t = 0; t < 8; ++t) a[8 + t] = p[8 + t];
    return a;
}

// B (32x16 bf16) from K-contiguous (transposed) weights: row = output column,
// ldk = K stride. lane<16: N=lane, K=kb..kb+15; lane>=16: N=lane-16, K=kb+16..31.
__device__ inline v16bf load_b_frag(const __bf16* __restrict__ colbase, size_t ldk,
                                    int kb, int lane) {
    const __bf16* p = colbase + (size_t)(lane & 15) * ldk + kb + ((lane < 16) ? 0 : 16);
    v16bf b;
    #pragma unroll
    for (int t = 0; t < 16; ++t) b[t] = p[t];
    return b;
}

// =====================================================================
// Fused SwiGLU GEMM: out_tile[16, D] = (silu(X@Wg) * (X@Wu)) @ Wd
// MODE 0: shared expert (contiguous tokens, Iext = SI, plain store)
// MODE 1: routed experts (gathered tokens, per-row combine weight, store to eo)
// Block = 256 threads = 8 wave32; wave w owns i-cols [w*16] of each 128-chunk
// and d-cols [w*128, w*128+128) of the output.
// =====================================================================
template <int MODE>
__global__ __launch_bounds__(256) void moe_swiglu_gemm(
    const __bf16* __restrict__ xb,
    const __bf16* __restrict__ wgt,   // [I or SI][D] per expert (gate^T)
    const __bf16* __restrict__ wut,   // [I or SI][D] per expert (up^T)
    const __bf16* __restrict__ wdt,   // [D][I or SI] per expert (down^T)
    const int* __restrict__ btok, const float* __restrict__ bw,
    const int* __restrict__ offsets,
    float* __restrict__ outp, int Iext) {
    constexpr int XS = cD + 8;        // padded LDS strides (bank-conflict-free b128)
    constexpr int HS = 128 + 8;
    __shared__ __bf16 xs[16 * XS];
    __shared__ __bf16 hs[16 * HS];
    __shared__ int   tok_s[16];
    __shared__ float w_s[16];

    int tid = threadIdx.x;
    int lane = tid & 31, wave = tid >> 5;

    int e = 0, tile = blockIdx.x, rstart = 0, nvalid = 16;
    if (MODE == 1) {
        e = blockIdx.x >> 8;
        tile = blockIdx.x & 255;
        rstart = offsets[e];
        int cnt = offsets[e + 1] - rstart;
        if (tile * 16 >= cnt) return;             // block-uniform early exit
        nvalid = min(16, cnt - tile * 16);
    }

    if (tid < 16) {
        int tok; float w = 1.f;
        if (MODE == 1) {
            if (tid < nvalid) { int gr = rstart + tile * 16 + tid; tok = btok[gr]; w = bw[gr]; }
            else tok = 0;
        } else {
            tok = tile * 16 + tid;
        }
        tok_s[tid] = tok; w_s[tid] = w;
    }
    __syncthreads();

    // Stage X tile [16, D] into LDS (async DMA if available, else b128 copies)
    {
        int r = tid >> 4, seg = tid & 15;
        const __bf16* src = xb + (size_t)tok_s[r] * cD;
        #pragma unroll
        for (int it = 0; it < 8; ++it) {
            int c = seg * 8 + it * 128;
#ifdef USE_ASYNC_LDS
            __builtin_amdgcn_global_load_async_to_lds_b128(
                to_global_v4(src + c), to_lds_v4(&xs[r * XS + c]), 0, 0);
#else
            *(v8bf*)&xs[r * XS + c] = *(const v8bf*)&src[c];
#endif
        }
    }
#ifdef USE_ASYNC_LDS
    __builtin_amdgcn_s_wait_asynccnt(0);
#endif
    __syncthreads();

    const __bf16* wg_e = wgt + (MODE == 1 ? (size_t)e * cI * cD : 0);
    const __bf16* wu_e = wut + (MODE == 1 ? (size_t)e * cI * cD : 0);
    const __bf16* wd_e = wdt + (MODE == 1 ? (size_t)e * cD * cI : 0);

    v8f zero = {};
    v8f dacc[8];
    #pragma unroll
    for (int s = 0; s < 8; ++s) dacc[s] = zero;

    int nchunks = Iext >> 7;
    for (int chunk = 0; chunk < nchunks; ++chunk) {
        int icol0 = chunk * 128 + wave * 16;
        const __bf16* gcol = wg_e + (size_t)icol0 * cD;
        const __bf16* ucol = wu_e + (size_t)icol0 * cD;
        if (chunk + 1 < nchunks)   // hint next chunk's weights toward L2
            __builtin_prefetch(wg_e + (size_t)(icol0 + 128) * cD + lane * 64, 0, 1);

        v8f g = zero, u = zero;
        #pragma unroll 4
        for (int kb = 0; kb < cD; kb += 32) {
            v16bf a  = load_a_frag(xs, XS, kb, lane);
            v16bf bg = load_b_frag(gcol, cD, kb, lane);
            v16bf bu = load_b_frag(ucol, cD, kb, lane);
            g = __builtin_amdgcn_wmma_f32_16x16x32_bf16(false, a, false, bg, (short)0, g, false, false);
            u = __builtin_amdgcn_wmma_f32_16x16x32_bf16(false, a, false, bu, (short)0, u, false, false);
        }

        // H = silu(g) * u  -> LDS (bf16).  sigmoid via v_exp + v_rcp (no IEEE div:
        // data is bf16-quantized anyway; the div_scale/div_fmas chain is wasted VALU)
        {
            int mb = (lane < 16) ? 0 : 8;
            int nc = lane & 15;
            #pragma unroll
            for (int r = 0; r < 8; ++r) {
                float gv = g[r];
                float sig = __builtin_amdgcn_rcpf(1.0f + __expf(-gv));
                float hv = gv * sig * u[r];
                hs[(mb + r) * HS + wave * 16 + nc] = (__bf16)hv;
            }
        }
        __syncthreads();

        // down: acc[16, wave's 128 d-cols] += H[16,128-chunk] @ Wd_chunk
        #pragma unroll
        for (int ks = 0; ks < 4; ++ks) {
            v16bf a = load_a_frag(hs, HS, ks * 32, lane);
            int kg = chunk * 128 + ks * 32;
            #pragma unroll
            for (int s = 0; s < 8; ++s) {
                const __bf16* dcol = wd_e + (size_t)(wave * 128 + s * 16) * Iext + kg;
                v16bf b = load_b_frag(dcol, (size_t)Iext, 0, lane);
                dacc[s] = __builtin_amdgcn_wmma_f32_16x16x32_bf16(false, a, false, b, (short)0, dacc[s], false, false);
            }
        }
        __syncthreads();   // hs reused next chunk
    }

    // Store output tile (C/D layout: VGPR r -> M = r + 8*(lane>=16), N = lane&15)
    {
        int mb = (lane < 16) ? 0 : 8;
        int nc = lane & 15;
        #pragma unroll
        for (int s = 0; s < 8; ++s) {
            int col = wave * 128 + s * 16 + nc;
            #pragma unroll
            for (int r = 0; r < 8; ++r) {
                int m = mb + r;
                if (MODE == 1) {
                    if (m < nvalid)
                        outp[(size_t)(rstart + tile * 16 + m) * cD + col] = dacc[s][r] * w_s[m];
                } else {
                    outp[(size_t)(tile * 16 + m) * cD + col] = dacc[s][r];
                }
            }
        }
    }
}

// =====================================================================
// Host launcher
// =====================================================================
extern "C" void kernel_launch(void* const* d_in, const int* in_sizes, int n_in,
                              void* d_out, int out_size, void* d_ws, size_t ws_size,
                              hipStream_t stream) {
    (void)in_sizes; (void)n_in; (void)out_size; (void)ws_size;
    const float* x   = (const float*)d_in[0];
    const float* Wr  = (const float*)d_in[1];
    const float* Wg  = (const float*)d_in[2];
    const float* Wu  = (const float*)d_in[3];
    const float* Wd  = (const float*)d_in[4];
    const float* Wsg = (const float*)d_in[5];
    const float* Wsu = (const float*)d_in[6];
    const float* Wsd = (const float*)d_in[7];

    float* out    = (float*)d_out;                 // [N, D]
    float* logits = out + (size_t)cN * cD;         // [N, E]

    // ---- workspace carve-up (256B aligned) ----
    char* p = (char*)d_ws;
    auto take = [&](size_t bytes) {
        void* r = (void*)p;
        p += (bytes + 255) & ~(size_t)255;
        return r;
    };
    __bf16* xb   = (__bf16*)take((size_t)cN * cD * 2);
    __bf16* wgt  = (__bf16*)take((size_t)cE * cI * cD * 2);
    __bf16* wut  = (__bf16*)take((size_t)cE * cI * cD * 2);
    __bf16* wdt  = (__bf16*)take((size_t)cE * cD * cI * 2);
    __bf16* wsgt = (__bf16*)take((size_t)cSI * cD * 2);
    __bf16* wsut = (__bf16*)take((size_t)cSI * cD * 2);
    __bf16* wsdt = (__bf16*)take((size_t)cD * cSI * 2);
    float*  eo   = (float*)take((size_t)cN * cK * cD * 4);   // per-assignment rows
    int*    assign_e = (int*)take((size_t)cN * cK * 4);
    float*  assign_w = (float*)take((size_t)cN * cK * 4);
    int*    btok     = (int*)take((size_t)cN * cK * 4);
    float*  bw       = (float*)take((size_t)cN * cK * 4);
    int*    posof    = (int*)take((size_t)cN * cK * 4);
    int*    counts   = (int*)take(64);
    int*    offsets  = (int*)take(128);
    int*    cursors  = (int*)take(64);

    dim3 tb(32, 8);

    init_kernel<<<1, 32, 0, stream>>>(counts);
    convert_x_kernel<<<(cN * cD / 4) / 256, 256, 0, stream>>>(x, xb);

    // weight transpose+convert: f32 [R][C] -> bf16 [C][R]
    transpose_cvt_kernel<<<dim3(cI / 32, cD / 32, cE), tb, 0, stream>>>(Wg, wgt, cD, cI);
    transpose_cvt_kernel<<<dim3(cI / 32, cD / 32, cE), tb, 0, stream>>>(Wu, wut, cD, cI);
    transpose_cvt_kernel<<<dim3(cD / 32, cI / 32, cE), tb, 0, stream>>>(Wd, wdt, cI, cD);
    transpose_cvt_kernel<<<dim3(cSI / 32, cD / 32, 1), tb, 0, stream>>>(Wsg, wsgt, cD, cSI);
    transpose_cvt_kernel<<<dim3(cSI / 32, cD / 32, 1), tb, 0, stream>>>(Wsu, wsut, cD, cSI);
    transpose_cvt_kernel<<<dim3(cD / 32, cSI / 32, 1), tb, 0, stream>>>(Wsd, wsdt, cSI, cD);

    router_kernel<<<cN / 8, 256, 0, stream>>>(x, Wr, logits);
    topk_kernel<<<cN / 256, 256, 0, stream>>>(logits, assign_e, assign_w, counts);
    scan_kernel<<<1, 32, 0, stream>>>(counts, offsets, cursors);
    scatter_kernel<<<(cN * cK) / 256, 256, 0, stream>>>(assign_e, assign_w, cursors,
                                                        btok, bw, posof);

    // shared expert writes out (fully initializes output region)
    moe_swiglu_gemm<0><<<cN / 16, 256, 0, stream>>>(xb, wsgt, wsut, wsdt,
                                                    nullptr, nullptr, nullptr, out, cSI);
    // routed experts -> per-assignment buffer (worst-case grid, cheap early exit)
    moe_swiglu_gemm<1><<<cE * 256, 256, 0, stream>>>(xb, wgt, wut, wdt,
                                                     btok, bw, offsets, eo, cI);
    // deterministic combine: out += eo[pos0] + eo[pos1]
    combine_kernel<<<(cN * cD / 4) / 256, 256, 0, stream>>>(out, eo, posof);
}
